// ShiftedWindowPatchAE_39204461478025
// MI455X (gfx1250) — matrix operand order repaired and checked by reading, not compiled, not run on hardware
//
#include <hip/hip_runtime.h>
#include <hip/hip_bf16.h>

// ---------------------------------------------------------------------------
// ShiftedWindowPatchAE forward for MI455X (gfx1250, wave32, WMMA).
// bf16 operands resident in HBM; async GLOBAL_LOAD_ASYNC_TO_LDS_B128 staging
// into double-buffered bf16 LDS tiles; v_wmma_f32_16x16x32_bf16 compute with
// fp32 accumulation; zero conversion VALU in the GEMM inner loop.
// ---------------------------------------------------------------------------

#define Bdim   64
#define Gdim   16
#define Ndim   256              // G*G tokens per image
#define Ddim   512
#define Tdim   (Bdim * Ndim)    // 16384 tokens
#define HIDdim 2048
#define Mdim   512              // memory slots per cell
#define Ktop   16

typedef __attribute__((ext_vector_type(16))) __bf16 bf16x16;
typedef __attribute__((ext_vector_type(8)))  __bf16 bf16x8;
typedef __attribute__((ext_vector_type(8)))  float  f32x8;
typedef __attribute__((ext_vector_type(4)))  int    v4i;

union FragBF16 { bf16x16 v; bf16x8 h[2]; };

#if defined(__has_builtin)
#if __has_builtin(__builtin_amdgcn_global_load_async_to_lds_b128) && \
    __has_builtin(__builtin_amdgcn_s_wait_asynccnt)
#define HAVE_ASYNC 1
#endif
#endif
#ifndef HAVE_ASYNC
#define HAVE_ASYNC 0
#endif

static __device__ inline f32x8 zero8() {
    f32x8 z = {0.f, 0.f, 0.f, 0.f, 0.f, 0.f, 0.f, 0.f};
    return z;
}

static __device__ inline float gelu_tanh(float x) {
    return 0.5f * x * (1.f + tanhf(0.7978845608028654f * (x + 0.044715f * x * x * x)));
}

// 16-byte global -> LDS copy: async DMA (ASYNCcnt) on CDNA5, sync fallback.
static __device__ inline void copy16(const void* g, void* l) {
#if HAVE_ASYNC
    __builtin_amdgcn_global_load_async_to_lds_b128(
        (__attribute__((address_space(1))) v4i*)g,
        (__attribute__((address_space(3))) v4i*)l, 0, 0);
#else
    *(v4i*)l = *(const v4i*)g;
#endif
}
static __device__ inline void wait_async_4() {
#if HAVE_ASYNC
    __builtin_amdgcn_s_wait_asynccnt(4);
#endif
}
static __device__ inline void wait_async_0() {
#if HAVE_ASYNC
    __builtin_amdgcn_s_wait_asynccnt(0);
#endif
}

// ---------------------------------------------------------------------------
// fp32 -> bf16 bulk conversion (n multiple of 8).
// ---------------------------------------------------------------------------
__global__ __launch_bounds__(256) void f32_to_bf16_kernel(
    const float* __restrict__ in, __bf16* __restrict__ out, long n)
{
    long i = ((long)blockIdx.x * 256 + threadIdx.x) * 8;
    if (i >= n) return;
    float4 a = *(const float4*)(in + i);
    float4 b = *(const float4*)(in + i + 4);
    bf16x8 v;
    v[0] = (__bf16)a.x; v[1] = (__bf16)a.y; v[2] = (__bf16)a.z; v[3] = (__bf16)a.w;
    v[4] = (__bf16)b.x; v[5] = (__bf16)b.y; v[6] = (__bf16)b.z; v[7] = (__bf16)b.w;
    *(bf16x8*)(out + i) = v;
}

// ---------------------------------------------------------------------------
// Weight transpose with bf16 output: out[c][r] = bf16(in[r][c]).
// ---------------------------------------------------------------------------
__global__ __launch_bounds__(256) void transpose_bf16_kernel(
    const float* __restrict__ in, __bf16* __restrict__ out, int R, int Cc)
{
    __shared__ float tile[32][33];
    int c0 = blockIdx.x * 32, r0 = blockIdx.y * 32;
    int tx = threadIdx.x & 31, ty = threadIdx.x >> 5;
    for (int i = ty; i < 32; i += 8)
        tile[i][tx] = in[(size_t)(r0 + i) * Cc + c0 + tx];
    __syncthreads();
    for (int i = ty; i < 32; i += 8)
        out[(size_t)(c0 + i) * R + r0 + tx] = (__bf16)tile[tx][i];
}

// ---------------------------------------------------------------------------
// Patch embedding: pad 28x28 -> 32x32, strided 2x2 conv, + bias + pos embed.
// ---------------------------------------------------------------------------
__global__ __launch_bounds__(256) void patch_embed_kernel(
    const float* __restrict__ x, const float* __restrict__ pw,
    const float* __restrict__ pb, const float* __restrict__ pos,
    float* __restrict__ out)
{
    int t = blockIdx.x;
    int b = t >> 8, n = t & 255;
    int gy = n >> 4, gx = n & 15;
    for (int d = threadIdx.x; d < Ddim; d += 256) {
        float acc = pb[d] + pos[n * Ddim + d];
        #pragma unroll
        for (int py = 0; py < 2; ++py)
            #pragma unroll
            for (int px = 0; px < 2; ++px) {
                int ih = gy * 2 + py - 2, iw = gx * 2 + px - 2;
                if (ih >= 0 && ih < 28 && iw >= 0 && iw < 28)
                    acc += x[(size_t)b * 784 + ih * 28 + iw] * pw[d * 4 + py * 2 + px];
            }
        out[(size_t)t * Ddim + d] = acc;
    }
}

// ---------------------------------------------------------------------------
// Windowed-token index -> original token index (cyclic shift + 4x4 windows).
// ---------------------------------------------------------------------------
__global__ void rowmap_kernel(int* __restrict__ map)
{
    int t = blockIdx.x * blockDim.x + threadIdx.x;
    if (t >= Tdim) return;
    int i    = t & 15;
    int wlin = t >> 4;
    int wx = wlin & 3, wy = (wlin >> 2) & 3, b = wlin >> 4;
    int iy = i >> 2, ix = i & 3;
    int gy = (wy * 4 + iy + 2) & 15;
    int gx = (wx * 4 + ix + 2) & 15;
    map[t] = b * Ndim + gy * Gdim + gx;
}

// ---------------------------------------------------------------------------
// LayerNorm over D=512 (fp32 in), bf16 out, optional input-row gather.
// ---------------------------------------------------------------------------
__global__ __launch_bounds__(256) void ln_bf16_kernel(
    const float* __restrict__ in, const int* __restrict__ map,
    const float* __restrict__ gs, const float* __restrict__ gb,
    __bf16* __restrict__ out)
{
    int row = blockIdx.x;
    int src = map ? map[row] : row;
    const float* p = in + (size_t)src * Ddim;
    int tid = threadIdx.x;
    float x0 = p[tid], x1 = p[tid + 256];
    __shared__ float rs[256], rq[256];
    rs[tid] = x0 + x1;
    rq[tid] = x0 * x0 + x1 * x1;
    __syncthreads();
    for (int s = 128; s > 0; s >>= 1) {
        if (tid < s) { rs[tid] += rs[tid + s]; rq[tid] += rq[tid + s]; }
        __syncthreads();
    }
    float mean = rs[0] * (1.f / Ddim);
    float var  = rq[0] * (1.f / Ddim) - mean * mean;
    float inv  = rsqrtf(var + 1e-5f);
    out[(size_t)row * Ddim + tid]       = (__bf16)((x0 - mean) * inv * gs[tid]       + gb[tid]);
    out[(size_t)row * Ddim + tid + 256] = (__bf16)((x1 - mean) * inv * gs[tid + 256] + gb[tid + 256]);
}

// ---------------------------------------------------------------------------
// Squared row norms (fp32 in).
// ---------------------------------------------------------------------------
__global__ __launch_bounds__(256) void sqnorm_kernel(
    const float* __restrict__ in, float* __restrict__ out)
{
    int row = blockIdx.x;
    const float* p = in + (size_t)row * Ddim;
    int tid = threadIdx.x;
    float x0 = p[tid], x1 = p[tid + 256];
    __shared__ float rq[256];
    rq[tid] = x0 * x0 + x1 * x1;
    __syncthreads();
    for (int s = 128; s > 0; s >>= 1) {
        if (tid < s) rq[tid] += rq[tid + s];
        __syncthreads();
    }
    if (tid == 0) out[row] = rq[0];
}

// ---------------------------------------------------------------------------
// bf16-WMMA GEMM (NT): out = act(A[M,K] * Bt[N,K]^T + bias) (+residual).
// 128 threads = 4 waves, 64x64 tile, BK=32, double-buffered bf16 LDS tiles
// staged with async b128 copies; fragments load directly (no converts).
// ---------------------------------------------------------------------------
__global__ __launch_bounds__(128) void wmma_gemm_bf16_kernel(
    const __bf16* __restrict__ A, int lda,
    const __bf16* __restrict__ Bt, int ldb,   // [N][K]
    const float* __restrict__ bias,
    float*  __restrict__ C,                   // fp32 out (or null)
    __bf16* __restrict__ Cb,                  // bf16 out (or null)
    int ldc,
    const float* __restrict__ Cres,
    const int* __restrict__ rowMap,
    int Kd, int doGelu)
{
    const int m0 = blockIdx.y * 64;
    const int n0 = blockIdx.x * 64;
    __shared__ __align__(16) __bf16 As[2][64][40];   // 80B row stride
    __shared__ __align__(16) __bf16 Bs[2][64][40];
    const int tid  = threadIdx.x;
    const int lane = tid & 31, wv = tid >> 5;
    const int rr = lane & 15, hi = lane >> 4;

    // 64x32 bf16 tile = 256 x 16B chunks; 2 A-chunks + 2 B-chunks per thread.
    auto stage = [&](int buf, int k0) {
        #pragma unroll
        for (int j = 0; j < 2; ++j) {
            int c = tid + j * 128;
            int r = c >> 2, q = (c & 3) * 8;
            copy16(&A [(size_t)(m0 + r) * lda + k0 + q], &As[buf][r][q]);
            copy16(&Bt[(size_t)(n0 + r) * ldb + k0 + q], &Bs[buf][r][q]);
        }
    };

    f32x8 acc[4];
    #pragma unroll
    for (int j = 0; j < 4; ++j) acc[j] = zero8();

    stage(0, 0);
    int buf = 0;
    for (int k0 = 0; k0 < Kd; k0 += 32) {
        if (k0 + 32 < Kd) { stage(buf ^ 1, k0 + 32); wait_async_4(); }
        else              { wait_async_0(); }
        __syncthreads();

        // A frag: lane rr = row; elems 0..7 <-> K=hi*8+e, 8..15 <-> 16+hi*8+e
        FragBF16 af;
        af.h[0] = *(const bf16x8*)&As[buf][wv * 16 + rr][hi * 8];
        af.h[1] = *(const bf16x8*)&As[buf][wv * 16 + rr][16 + hi * 8];
        #pragma unroll
        for (int jt = 0; jt < 4; ++jt) {
            // B frag: lane rr = col; elems 0..7 <-> K=hi*8+e, 8..15 <-> 16+hi*8+e
            FragBF16 bf;
            bf.h[0] = *(const bf16x8*)&Bs[buf][jt * 16 + rr][hi * 8];
            bf.h[1] = *(const bf16x8*)&Bs[buf][jt * 16 + rr][16 + hi * 8];
            acc[jt] = __builtin_amdgcn_wmma_f32_16x16x32_bf16(
                false, af.v, false, bf.v, (short)0, acc[jt], false, false);
        }
        __syncthreads();
        buf ^= 1;
    }

    // Epilogue: C layout -> lane rr = col, row = hi*8 + vgpr index
    #pragma unroll
    for (int jt = 0; jt < 4; ++jt) {
        int gn = n0 + jt * 16 + rr;
        float bv = bias ? bias[gn] : 0.f;
        #pragma unroll
        for (int e = 0; e < 8; ++e) {
            int gm = m0 + wv * 16 + hi * 8 + e;
            float v = acc[jt][e] + bv;
            if (doGelu) v = gelu_tanh(v);
            int orow = rowMap ? rowMap[gm] : gm;
            size_t idx = (size_t)orow * ldc + gn;
            if (Cres) v += Cres[idx];
            if (C)  C[idx]  = v;
            if (Cb) Cb[idx] = (__bf16)v;
        }
    }
}

// ---------------------------------------------------------------------------
// Shifted-window attention: one wave per 16-token window, bf16 in/out.
// QK^T via WMMA (fragments load straight from global bf16 qkv), softmax via
// 16-lane shuffles, P transposed through LDS, P*V via zero-padded WMMA.
// ---------------------------------------------------------------------------
__global__ __launch_bounds__(128) void attn_kernel(
    const __bf16* __restrict__ qkv, __bf16* __restrict__ ho)
{
    const int lane = threadIdx.x & 31, wv = threadIdx.x >> 5;
    const int rr = lane & 15, hi = lane >> 4;
    const int w = blockIdx.x * 4 + wv;                 // 1024 windows
    const __bf16* Qp = qkv + (size_t)w * 16 * 1536;
    const __bf16* Kp = Qp + 512;
    const __bf16* Vp = Qp + 1024;
    __shared__ float Ps[4][16][17];

    f32x8 sc = zero8();
    for (int k0 = 0; k0 < Ddim; k0 += 32) {
        const __bf16* qr = Qp + rr * 1536 + k0;
        const __bf16* kr = Kp + rr * 1536 + k0;
        if (k0 + 32 < Ddim) __builtin_prefetch(qr + 32, 0, 0);
        FragBF16 a, b;
        a.h[0] = *(const bf16x8*)(qr + hi * 8);
        a.h[1] = *(const bf16x8*)(qr + 16 + hi * 8);
        b.h[0] = *(const bf16x8*)(kr + hi * 8);
        b.h[1] = *(const bf16x8*)(kr + 16 + hi * 8);
        sc = __builtin_amdgcn_wmma_f32_16x16x32_bf16(
            false, a.v, false, b.v, (short)0, sc, false, false);
    }

    const float scale = 0.04419417382415922f;          // 1/sqrt(512)
    float p[8];
    #pragma unroll
    for (int e = 0; e < 8; ++e) {
        float v = sc[e] * scale;
        float mx = v;
        for (int off = 1; off < 16; off <<= 1) mx = fmaxf(mx, __shfl_xor(mx, off, 32));
        float ex = __expf(v - mx);
        float sm = ex;
        for (int off = 1; off < 16; off <<= 1) sm += __shfl_xor(sm, off, 32);
        p[e] = ex / sm;
    }
    #pragma unroll
    for (int e = 0; e < 8; ++e) Ps[wv][hi * 8 + e][rr] = p[e];
    __syncthreads();

    FragBF16 pa;     // A frag of P: K = key index (16..31 zero-padded)
    #pragma unroll
    for (int e = 0; e < 8; ++e) {
        pa.v[e]     = (__bf16)Ps[wv][rr][hi * 8 + e];
        pa.v[8 + e] = (__bf16)0.0f;
    }
    for (int d0 = 0; d0 < Ddim; d0 += 16) {
        FragBF16 vb; // B frag of V: elems 0..7 <-> key hi*8+e, 8..15 -> padded zeros
        #pragma unroll
        for (int e = 0; e < 16; ++e)
            vb.v[e] = (e >= 8) ? (__bf16)0.0f : Vp[(hi * 8 + e) * 1536 + d0 + rr];
        f32x8 o = zero8();
        o = __builtin_amdgcn_wmma_f32_16x16x32_bf16(
            false, pa.v, false, vb.v, (short)0, o, false, false);
        #pragma unroll
        for (int e = 0; e < 8; ++e)
            ho[((size_t)w * 16 + hi * 8 + e) * Ddim + d0 + rr] = (__bf16)o[e];
    }
}

// ---------------------------------------------------------------------------
// Batched memory-distance GEMM per grid cell (bf16 operands):
// dist[b, m] = sqrt(max(||x||^2 + ||k||^2 - 2 x.k, 1e-12)).
// ---------------------------------------------------------------------------
__global__ __launch_bounds__(128) void wmma_dist_kernel(
    const __bf16* __restrict__ X,    // [B, N, D] bf16
    const __bf16* __restrict__ Keys, // [N, M, D] bf16
    const float* __restrict__ qn,    // [B*N]
    const float* __restrict__ kn,    // [N*M]
    float* __restrict__ Dist)        // [B*N, M]
{
    const int cell = blockIdx.z;
    const int n0 = blockIdx.x * 64;
    const __bf16* A  = X + (size_t)cell * Ddim;              // row b, stride N*D
    const __bf16* Bm = Keys + (size_t)cell * Mdim * Ddim;    // [M][D]
    __shared__ __align__(16) __bf16 As[2][64][40];
    __shared__ __align__(16) __bf16 Bs[2][64][40];
    const int tid = threadIdx.x;
    const int lane = tid & 31, wv = tid >> 5;
    const int rr = lane & 15, hi = lane >> 4;

    auto stage = [&](int buf, int k0) {
        #pragma unroll
        for (int j = 0; j < 2; ++j) {
            int c = tid + j * 128;
            int r = c >> 2, q = (c & 3) * 8;
            copy16(&A [(size_t)r * (Ndim * Ddim) + k0 + q], &As[buf][r][q]);
            copy16(&Bm[(size_t)(n0 + r) * Ddim + k0 + q],   &Bs[buf][r][q]);
        }
    };

    f32x8 acc[4];
    #pragma unroll
    for (int j = 0; j < 4; ++j) acc[j] = zero8();

    stage(0, 0);
    int buf = 0;
    for (int k0 = 0; k0 < Ddim; k0 += 32) {
        if (k0 + 32 < Ddim) { stage(buf ^ 1, k0 + 32); wait_async_4(); }
        else                { wait_async_0(); }
        __syncthreads();
        FragBF16 af;
        af.h[0] = *(const bf16x8*)&As[buf][wv * 16 + rr][hi * 8];
        af.h[1] = *(const bf16x8*)&As[buf][wv * 16 + rr][16 + hi * 8];
        #pragma unroll
        for (int jt = 0; jt < 4; ++jt) {
            FragBF16 bf;
            bf.h[0] = *(const bf16x8*)&Bs[buf][jt * 16 + rr][hi * 8];
            bf.h[1] = *(const bf16x8*)&Bs[buf][jt * 16 + rr][16 + hi * 8];
            acc[jt] = __builtin_amdgcn_wmma_f32_16x16x32_bf16(
                false, af.v, false, bf.v, (short)0, acc[jt], false, false);
        }
        __syncthreads();
        buf ^= 1;
    }
    #pragma unroll
    for (int jt = 0; jt < 4; ++jt) {
        int gn = n0 + jt * 16 + rr;                 // m
        float knv = kn[cell * Mdim + gn];
        #pragma unroll
        for (int e = 0; e < 8; ++e) {
            int gm = wv * 16 + hi * 8 + e;          // b
            int token = gm * Ndim + cell;
            float d2 = qn[token] + knv - 2.f * acc[jt][e];
            Dist[(size_t)token * Mdim + gn] = sqrtf(fmaxf(d2, 1e-12f));
        }
    }
}

// ---------------------------------------------------------------------------
// Top-16 smallest distances + softmax(-dist) + weighted value sum.
// ---------------------------------------------------------------------------
__global__ __launch_bounds__(128) void topk_mem_kernel(
    const float* __restrict__ dist, const float* __restrict__ mvals,
    float* __restrict__ out)
{
    int t = blockIdx.x;
    int n = t & 255;
    int tid = threadIdx.x;
    __shared__ float dv[512];
    __shared__ float redv[128];
    __shared__ int   redi[128];
    __shared__ float topv[Ktop];
    __shared__ int   topi[Ktop];
    __shared__ float wts[Ktop];
    for (int i = tid; i < Mdim; i += 128) dv[i] = dist[(size_t)t * Mdim + i];
    __syncthreads();

    for (int k = 0; k < Ktop; ++k) {
        float bv = 3.0e38f; int bi = 0x7fffffff;
        for (int i = tid; i < Mdim; i += 128) {
            float v = dv[i];
            if (v < bv || (v == bv && i < bi)) { bv = v; bi = i; }
        }
        redv[tid] = bv; redi[tid] = bi;
        __syncthreads();
        for (int s = 64; s > 0; s >>= 1) {
            if (tid < s) {
                float ov = redv[tid + s]; int oi = redi[tid + s];
                if (ov < redv[tid] || (ov == redv[tid] && oi < redi[tid])) {
                    redv[tid] = ov; redi[tid] = oi;
                }
            }
            __syncthreads();
        }
        if (tid == 0) { topv[k] = redv[0]; topi[k] = redi[0]; dv[redi[0]] = 3.0e38f; }
        __syncthreads();
    }

    if (tid == 0) {
        float mx = -3.0e38f;
        for (int k = 0; k < Ktop; ++k) mx = fmaxf(mx, -topv[k]);
        float s = 0.f;
        for (int k = 0; k < Ktop; ++k) { float w = __expf(-topv[k] - mx); wts[k] = w; s += w; }
        float inv = 1.f / s;
        for (int k = 0; k < Ktop; ++k) wts[k] *= inv;
    }
    __syncthreads();

    for (int d = tid; d < Ddim; d += 128) {
        float a = 0.f;
        #pragma unroll
        for (int k = 0; k < Ktop; ++k)
            a += wts[k] * mvals[((size_t)n * Mdim + topi[k]) * Ddim + d];
        out[(size_t)t * Ddim + d] = a;
    }
}

// ---------------------------------------------------------------------------
// Decoder: token 512 -> 4 pixels, fold to image, crop to 28x28.
// ---------------------------------------------------------------------------
__global__ __launch_bounds__(256) void decoder_kernel(
    const float* __restrict__ xin, const float* __restrict__ dw,
    const float* __restrict__ db, float* __restrict__ img)
{
    int t = blockIdx.x;
    int b = t >> 8, n = t & 255;
    int gy = n >> 4, gx = n & 15;
    int tid = threadIdx.x;
    float a0 = 0, a1 = 0, a2 = 0, a3 = 0;
    for (int d = tid; d < Ddim; d += 256) {
        float xv = xin[(size_t)t * Ddim + d];
        a0 += xv * dw[d * 4 + 0];
        a1 += xv * dw[d * 4 + 1];
        a2 += xv * dw[d * 4 + 2];
        a3 += xv * dw[d * 4 + 3];
    }
    __shared__ float red[4][256];
    red[0][tid] = a0; red[1][tid] = a1; red[2][tid] = a2; red[3][tid] = a3;
    __syncthreads();
    for (int s = 128; s > 0; s >>= 1) {
        if (tid < s) {
            #pragma unroll
            for (int j = 0; j < 4; ++j) red[j][tid] += red[j][tid + s];
        }
        __syncthreads();
    }
    if (tid < 4) {
        int py = tid >> 1, px = tid & 1;
        int r = gy * 2 + py - 2, c = gx * 2 + px - 2;
        if (r >= 0 && r < 28 && c >= 0 && c < 28)
            img[(size_t)b * 784 + r * 28 + c] = red[tid][0] + db[tid];
    }
}

// ---------------------------------------------------------------------------
// Host orchestration
// ---------------------------------------------------------------------------
extern "C" void kernel_launch(void* const* d_in, const int* in_sizes, int n_in,
                              void* d_out, int out_size, void* d_ws, size_t ws_size,
                              hipStream_t stream)
{
    (void)in_sizes; (void)n_in; (void)out_size; (void)ws_size;
    const float* x_in    = (const float*)d_in[0];
    const float* patch_w = (const float*)d_in[1];
    const float* patch_b = (const float*)d_in[2];
    const float* pos     = (const float*)d_in[3];
    const float* ln1_s   = (const float*)d_in[4];
    const float* ln1_b   = (const float*)d_in[5];
    const float* wqkv    = (const float*)d_in[6];
    const float* bqkv    = (const float*)d_in[7];
    const float* wproj   = (const float*)d_in[8];
    const float* bproj   = (const float*)d_in[9];
    const float* ln2_s   = (const float*)d_in[10];
    const float* ln2_b   = (const float*)d_in[11];
    const float* w1      = (const float*)d_in[12];
    const float* b1      = (const float*)d_in[13];
    const float* w2      = (const float*)d_in[14];
    const float* b2      = (const float*)d_in[15];
    const float* memk    = (const float*)d_in[16];
    const float* memv    = (const float*)d_in[17];
    const float* dec_w   = (const float*)d_in[18];
    const float* dec_b   = (const float*)d_in[19];

    char* ws = (char*)d_ws;
    size_t off = 0;
    auto alloc = [&](size_t bytes) -> void* {
        void* p = ws + off;
        off += (bytes + 255) & ~(size_t)255;
        return p;
    };
    float*  x      = (float*) alloc((size_t)Tdim * Ddim * 4);      // residual (fp32)
    __bf16* hb     = (__bf16*)alloc((size_t)Tdim * Ddim * 2);      // LN out
    __bf16* qkvb   = (__bf16*)alloc((size_t)Tdim * 3 * Ddim * 2);
    __bf16* hob    = (__bf16*)alloc((size_t)Tdim * Ddim * 2);      // attn out
    __bf16* hidb   = (__bf16*)alloc((size_t)Tdim * HIDdim * 2);
    __bf16* xb     = (__bf16*)alloc((size_t)Tdim * Ddim * 2);      // x in bf16
    __bf16* memkb  = (__bf16*)alloc((size_t)Ndim * Mdim * Ddim * 2);
    float*  dist   = (float*) alloc((size_t)Tdim * Mdim * 4);
    float*  qn     = (float*) alloc((size_t)Tdim * 4);
    float*  kn     = (float*) alloc((size_t)Ndim * Mdim * 4);
    int*    rmap   = (int*)   alloc((size_t)Tdim * 4);
    float*  xm     = (float*) alloc((size_t)Tdim * Ddim * 4);      // memory output
    __bf16* wqkvT  = (__bf16*)alloc((size_t)Ddim * 3 * Ddim * 2);
    __bf16* wprojT = (__bf16*)alloc((size_t)Ddim * Ddim * 2);
    __bf16* w1T    = (__bf16*)alloc((size_t)Ddim * HIDdim * 2);
    __bf16* w2T    = (__bf16*)alloc((size_t)HIDdim * Ddim * 2);

    // 0. weight transposes -> bf16 NT layout; mem_keys -> bf16
    transpose_bf16_kernel<<<dim3(1536 / 32, Ddim / 32), 256, 0, stream>>>(wqkv, wqkvT, Ddim, 1536);
    transpose_bf16_kernel<<<dim3(Ddim / 32, Ddim / 32), 256, 0, stream>>>(wproj, wprojT, Ddim, Ddim);
    transpose_bf16_kernel<<<dim3(HIDdim / 32, Ddim / 32), 256, 0, stream>>>(w1, w1T, Ddim, HIDdim);
    transpose_bf16_kernel<<<dim3(Ddim / 32, HIDdim / 32), 256, 0, stream>>>(w2, w2T, HIDdim, Ddim);
    {
        long nk = (long)Ndim * Mdim * Ddim;
        f32_to_bf16_kernel<<<(int)((nk / 8 + 255) / 256), 256, 0, stream>>>(memk, memkb, nk);
    }

    // 1. patch embed + pos
    patch_embed_kernel<<<Tdim, 256, 0, stream>>>(x_in, patch_w, patch_b, pos, x);
    // 2. shift+window permutation map
    rowmap_kernel<<<Tdim / 256, 256, 0, stream>>>(rmap);
    // 3. LN1 gathered into windowed order (bf16 out)
    ln_bf16_kernel<<<Tdim, 256, 0, stream>>>(x, rmap, ln1_s, ln1_b, hb);
    // 4. QKV GEMM -> bf16
    wmma_gemm_bf16_kernel<<<dim3(1536 / 64, Tdim / 64), 128, 0, stream>>>(
        hb, Ddim, wqkvT, Ddim, bqkv, nullptr, qkvb, 1536, nullptr, nullptr, Ddim, 0);
    // 5. window attention (bf16 in/out)
    attn_kernel<<<Tdim / 64, 128, 0, stream>>>(qkvb, hob);
    // 6. proj GEMM + un-window/un-shift scatter + residual into fp32 x
    wmma_gemm_bf16_kernel<<<dim3(Ddim / 64, Tdim / 64), 128, 0, stream>>>(
        hob, Ddim, wprojT, Ddim, bproj, x, nullptr, Ddim, x, rmap, Ddim, 0);
    // 7. LN2 (bf16 out)
    ln_bf16_kernel<<<Tdim, 256, 0, stream>>>(x, nullptr, ln2_s, ln2_b, hb);
    // 8. MLP1 GEMM + GELU -> bf16
    wmma_gemm_bf16_kernel<<<dim3(HIDdim / 64, Tdim / 64), 128, 0, stream>>>(
        hb, Ddim, w1T, Ddim, b1, nullptr, hidb, HIDdim, nullptr, nullptr, Ddim, 1);
    // 9. MLP2 GEMM + residual -> fp32 x
    wmma_gemm_bf16_kernel<<<dim3(Ddim / 64, Tdim / 64), 128, 0, stream>>>(
        hidb, HIDdim, w2T, HIDdim, b2, x, nullptr, Ddim, x, nullptr, HIDdim, 0);
    // 10. norms for cdist + bf16 copy of x
    sqnorm_kernel<<<Tdim, 256, 0, stream>>>(x, qn);
    sqnorm_kernel<<<Ndim * Mdim, 256, 0, stream>>>(memk, kn);
    f32_to_bf16_kernel<<<(int)(((long)Tdim * Ddim / 8 + 255) / 256), 256, 0, stream>>>(
        x, xb, (long)Tdim * Ddim);
    // 11. batched distance GEMM per grid cell
    wmma_dist_kernel<<<dim3(Mdim / 64, 1, Ndim), 128, 0, stream>>>(
        xb, memkb, qn, kn, dist);
    // 12. top-16 + softmax + weighted value sum -> xm
    topk_mem_kernel<<<Tdim, 128, 0, stream>>>(dist, memv, xm);
    // 13. decoder + fold + crop
    decoder_kernel<<<Tdim, 256, 0, stream>>>(xm, dec_w, dec_b, (float*)d_out);
}
